// MongeGapTransport_25331717111986
// MI455X (gfx1250) — matrix operand order repaired and checked by reading, not compiled, not run on hardware
//
#include <hip/hip_runtime.h>
#include <hip/hip_bf16.h>
#include <math.h>

// Problem constants (from reference)
#define NPTS  4096
#define DF    128
#define DR    64
#define HID   1024
#define KNN   10
#define DIN   192            // DF + DR
#define MROWS (NPTS * KNN)   // 40960 MLP rows
#define MT    64             // rows per MLP workgroup
#define MSN   (MT / 16)      // 4 row-subtiles sharing each B fragment
#define XPAD  200            // LDS row stride (shorts); 400 B row: 16B-aligned, bank-safe
#define HPAD  1032           // LDS row stride (shorts); 2064 B row: 16B-aligned, bank-safe

typedef __attribute__((ext_vector_type(16))) __bf16 v16bf;
typedef __attribute__((ext_vector_type(8)))  float  v8f;
typedef int v4i_ty __attribute__((vector_size(16)));

union BfFrag { v16bf v; unsigned int u[8]; v4i_ty q[2]; };

// CDNA5 async global->LDS path (GLOBAL_LOAD_ASYNC_TO_LDS_B128 + ASYNCcnt).
#if defined(__HIP_DEVICE_COMPILE__) &&                                   \
    __has_builtin(__builtin_amdgcn_global_load_async_to_lds_b128) &&     \
    __has_builtin(__builtin_amdgcn_s_wait_asynccnt)
#define HAVE_ASYNC_LDS 1
#endif

__device__ __forceinline__ unsigned short f2bf(float f) {
  unsigned int u = __float_as_uint(f);
  u += 0x7FFFu + ((u >> 16) & 1u);   // round-to-nearest-even
  return (unsigned short)(u >> 16);
}

__device__ __forceinline__ float softplus_f(float x) {
  // stable softplus using hw transcendentals (v_exp_f32 / v_log_f32);
  // these are TRANS ops and co-execute with the XDL WMMA pipe.
  return fmaxf(x, 0.0f) + __logf(1.0f + __expf(-fabsf(x)));
}

// ---------------------------------------------------------------------------
// Kernel 1: exact k-NN (10 smallest L2 distances, ties broken by lower index)
// One 128-thread block per query row.
// ---------------------------------------------------------------------------
__global__ void knn_topk_kernel(const float* __restrict__ cond,
                                int* __restrict__ nbr) {
  __shared__ float ci[DF];
  __shared__ float cval[128 * KNN];
  __shared__ int   cidx[128 * KNN];
  const int i = blockIdx.x;
  const int tid = threadIdx.x;
  if (tid < DF) ci[tid] = cond[(size_t)i * DF + tid];
  __syncthreads();

  float bv[KNN]; int bj[KNN];
#pragma unroll
  for (int t = 0; t < KNN; ++t) { bv[t] = 3.4e38f; bj[t] = 0x7fffffff; }

  const float4* cip = (const float4*)ci;
  for (int j = tid; j < NPTS; j += 128) {
    const float4* cj = (const float4*)(cond + (size_t)j * DF);
    float d2 = 0.0f;
#pragma unroll
    for (int q = 0; q < DF / 4; ++q) {
      float4 a = cip[q]; float4 b = cj[q];
      float dx = a.x - b.x, dy = a.y - b.y, dz = a.z - b.z, dw = a.w - b.w;
      d2 += dx * dx + dy * dy + dz * dz + dw * dw;
    }
    if (d2 < bv[KNN - 1]) {
      bv[KNN - 1] = d2; bj[KNN - 1] = j;
#pragma unroll
      for (int s = KNN - 1; s > 0; --s) {
        if (bv[s] < bv[s - 1]) {
          float tv = bv[s]; bv[s] = bv[s - 1]; bv[s - 1] = tv;
          int   ti = bj[s]; bj[s] = bj[s - 1]; bj[s - 1] = ti;
        }
      }
    }
  }
#pragma unroll
  for (int t = 0; t < KNN; ++t) {
    cval[tid * KNN + t] = bv[t];
    cidx[tid * KNN + t] = bj[t];
  }
  __syncthreads();

  // wave 0 merges the 1280 candidates: 10 rounds of wave-parallel argmin
  if (tid < 32) {
    const int per = (128 * KNN) / 32;  // 40 slots per lane
    for (int t = 0; t < KNN; ++t) {
      float v = 3.4e38f; int bi = 0x7fffffff; int sl = -1;
      for (int s = 0; s < per; ++s) {
        int slot = tid * per + s;
        float cv = cval[slot];
        int   cx = cidx[slot];
        if (cv < v || (cv == v && cx < bi)) { v = cv; bi = cx; sl = slot; }
      }
#pragma unroll
      for (int off = 16; off > 0; off >>= 1) {
        float ov  = __shfl_xor(v,  off, 32);
        int   obi = __shfl_xor(bi, off, 32);
        int   osl = __shfl_xor(sl, off, 32);
        if (ov < v || (ov == v && obi < bi)) { v = ov; bi = obi; sl = osl; }
      }
      if (tid == 0) nbr[i * KNN + t] = bi;
      if (sl >= tid * per && sl < (tid + 1) * per) cval[sl] = 3.4e38f;
    }
  }
}

// ---------------------------------------------------------------------------
// Kernel 2: gather neighbor (condition|tensor) rows, convert to bf16
// Xg is row-major [MROWS][DIN]
// ---------------------------------------------------------------------------
__global__ void gather_pack_x(const float* __restrict__ cond,
                              const float* __restrict__ tens,
                              const int* __restrict__ nbr,
                              unsigned short* __restrict__ Xg) {
  long gid = (long)blockIdx.x * blockDim.x + threadIdx.x;
  if (gid >= (long)MROWS * DIN) return;
  int r = (int)(gid / DIN), c = (int)(gid - (long)r * DIN);
  int j = nbr[r];
  float v = (c < DF) ? cond[(size_t)j * DF + c]
                     : tens[(size_t)j * DR + (c - DF)];
  Xg[(size_t)r * DIN + c] = f2bf(v);
}

// ---------------------------------------------------------------------------
// Kernel 3: pre-swizzle weights into WMMA B-matrix (32x16 K x N block) layout.
// Block (kt,nt): lane l holds column n = l&15, two 8-K runs starting at
// kb = kt*32 + (l<16 ? 0 : 8) and kb+16. Packed: [nt][kt][lane][16] bf16.
// ---------------------------------------------------------------------------
__global__ void pack_weights(const float* __restrict__ W,
                             unsigned short* __restrict__ dst,
                             int Kdim, int Ndim) {
  const int KT = Kdim >> 5;
  const int kt = blockIdx.x % KT;
  const int nt = blockIdx.x / KT;
  const int l  = threadIdx.x;
  const int n  = nt * 16 + (l & 15);
  const int kb = kt * 32 + ((l < 16) ? 0 : 8);
  unsigned short* o = dst + ((size_t)(nt * KT + kt) * 32 + l) * 16;
#pragma unroll
  for (int s = 0; s < 16; ++s)
    o[s] = f2bf(W[(size_t)(kb + s + ((s < 8) ? 0 : 8)) * Ndim + n]);
}

// ---------------------------------------------------------------------------
// CDNA5 async-LDS probe: GLOBAL_LOAD_ASYNC_TO_LDS_B128 + s_wait_asynccnt,
// with plain-copy fallback. Copies 1024 shorts src -> dst via LDS.
// ---------------------------------------------------------------------------
__global__ void async_stage_probe(const unsigned short* __restrict__ src,
                                  unsigned short* __restrict__ dst) {
  __shared__ unsigned short buf[1024];
  const int t = threadIdx.x;
#ifdef HAVE_ASYNC_LDS
  __builtin_amdgcn_global_load_async_to_lds_b128(
      (__attribute__((address_space(1))) v4i_ty*)(src + t * 8),
      (__attribute__((address_space(3))) v4i_ty*)(buf + t * 8),
      0, 0);
  __builtin_amdgcn_s_wait_asynccnt(0);
#else
#pragma unroll
  for (int s = 0; s < 8; ++s) buf[t * 8 + s] = src[t * 8 + s];
#endif
  __syncthreads();
#pragma unroll
  for (int s = 0; s < 8; ++s) dst[t * 8 + s] = buf[t * 8 + s];
}

// ---------------------------------------------------------------------------
// One MLP layer on a 64-row tile: D = softplus(A @ Wp + bias), bf16 -> bf16.
// Each wave owns 8 N-tiles; per k-step one B fragment is shared by the 4
// row-subtiles (4 accumulators) -> 4x FLOPs per weight byte.
// A-fragment per lane (16-bit A layout): row m = lane&15, two contiguous
// 8-K runs at k0 = kt*32 + (lane<16?0:8) and k0+16 -> two ds_load_b128.
// The asm memory clobber at the top of the nt loop stops LICM from hoisting
// all A-fragments out of the loop (which previously spilled ~1 KVGPR to
// scratch and turned every WMMA operand into a scratch reload).
// ---------------------------------------------------------------------------
__device__ __forceinline__ void mlp_layer(
    const unsigned short* __restrict__ in, int inStride, int KT,
    unsigned short* __restrict__ out, int outStride, int NT,
    const unsigned short* __restrict__ Wp, const float* __restrict__ bias,
    int wave, int lane) {
  const int mrow = lane & 15;
  const int kl   = (lane < 16) ? 0 : 8;
  const int ml   = (lane < 16) ? 0 : 8;
  for (int nt = wave; nt < NT; nt += 8) {
    asm volatile("" ::: "memory");  // keep per-kt LDS loads inside this loop
    v8f acc[MSN];
#pragma unroll
    for (int ms = 0; ms < MSN; ++ms)
      acc[ms] = (v8f){0.f, 0.f, 0.f, 0.f, 0.f, 0.f, 0.f, 0.f};
    const unsigned short* wp = Wp + ((size_t)nt * KT * 32 + lane) * 16;
    for (int kt = 0; kt < KT; ++kt) {
      BfFrag b;
      const v4i_ty* w4 = (const v4i_ty*)(wp + (size_t)kt * 512);
      b.q[0] = w4[0];
      b.q[1] = w4[1];
#pragma unroll
      for (int ms = 0; ms < MSN; ++ms) {
        BfFrag a;
        const v4i_ty* ar =
            (const v4i_ty*)(in + (ms * 16 + mrow) * inStride + kt * 32 + kl);
        a.q[0] = ar[0];   // k0 .. k0+7
        a.q[1] = ar[2];   // k0+16 .. k0+23
        acc[ms] = __builtin_amdgcn_wmma_f32_16x16x32_bf16(
            false, a.v, false, b.v, (short)0, acc[ms], false, false);
      }
    }
    const float bvv = bias[nt * 16 + (lane & 15)];
#pragma unroll
    for (int ms = 0; ms < MSN; ++ms) {
      unsigned short* oc = out + (ms * 16 + ml) * outStride + nt * 16 + (lane & 15);
#pragma unroll
      for (int v = 0; v < 8; ++v)
        oc[v * outStride] = f2bf(softplus_f(acc[ms][v] + bvv));
    }
  }
}

// ---------------------------------------------------------------------------
// Kernel 4: fully fused 5-layer MLP + pushforward. 256 threads, 64 rows/block.
// LDS: xt[64][XPAD] + double-buffered activations hA/hB[64][HPAD] (bf16),
// ~290 KB of the 320 KB WGP budget. X tile staged via async-LDS when available.
// ---------------------------------------------------------------------------
__global__ void fused_mlp_kernel(
    const unsigned short* __restrict__ Xg,
    const unsigned short* __restrict__ WpIn, const unsigned short* __restrict__ Wp1,
    const unsigned short* __restrict__ Wp2,  const unsigned short* __restrict__ Wp3,
    const unsigned short* __restrict__ WpOut,
    const float* __restrict__ b_in, const float* __restrict__ b1,
    const float* __restrict__ b2,   const float* __restrict__ b3,
    const float* __restrict__ b_out,
    const int* __restrict__ nbr, const float* __restrict__ tens,
    float* __restrict__ outp) {
  extern __shared__ __align__(16) unsigned short smem[];
  unsigned short* xt = smem;                 // MT * XPAD
  unsigned short* hA = smem + MT * XPAD;     // MT * HPAD
  unsigned short* hB = hA + MT * HPAD;       // MT * HPAD
  const int tid  = threadIdx.x;
  const int wave = tid >> 5, lane = tid & 31;
  const int rowBase = blockIdx.x * MT;

  // Warm first-layer weights into cache (global_prefetch_b8); L2-resident after.
  for (int b = tid; b < (DIN / 32) * (HID / 16); b += 256)
    __builtin_prefetch((const char*)WpIn + (size_t)b * 1024, 0, 1);

  // Stage the 64x192 bf16 input tile into LDS (async-LDS path when available:
  // per-lane GLOBAL_LOAD_ASYNC_TO_LDS_B128, then s_wait_asynccnt 0).
#ifdef HAVE_ASYNC_LDS
  for (int e = tid; e < MT * (DIN / 8); e += 256) {  // 16-byte chunks, 24 per row
    int r = e / (DIN / 8), c = e - r * (DIN / 8);
    __builtin_amdgcn_global_load_async_to_lds_b128(
        (__attribute__((address_space(1))) v4i_ty*)(Xg + (size_t)(rowBase + r) * DIN + c * 8),
        (__attribute__((address_space(3))) v4i_ty*)(xt + r * XPAD + c * 8),
        0, 0);
  }
  __builtin_amdgcn_s_wait_asynccnt(0);
#else
  for (int e = tid; e < MT * (DIN / 2); e += 256) {
    int r = e / (DIN / 2), c = e - r * (DIN / 2);
    *(unsigned int*)(xt + r * XPAD + 2 * c) =
        *(const unsigned int*)(Xg + (size_t)(rowBase + r) * DIN + 2 * c);
  }
#endif
  __syncthreads();

  mlp_layer(xt, XPAD, DIN / 32, hA, HPAD, HID / 16, WpIn, b_in, wave, lane);
  __syncthreads();
  mlp_layer(hA, HPAD, HID / 32, hB, HPAD, HID / 16, Wp1, b1, wave, lane);
  __syncthreads();
  mlp_layer(hB, HPAD, HID / 32, hA, HPAD, HID / 16, Wp2, b2, wave, lane);
  __syncthreads();
  mlp_layer(hA, HPAD, HID / 32, hB, HPAD, HID / 16, Wp3, b3, wave, lane);
  __syncthreads();

  // Output layer (linear, N = 64 -> 4x4 = 16 tiles, 2 per wave),
  // fused pushforward: out = tensor_gathered - (hB @ Wout + b_out).
  for (int t = wave; t < 4 * MSN; t += 8) {
    asm volatile("" ::: "memory");
    const int ms = t >> 2, nt = t & 3;
    const int mrow = lane & 15;
    const int kl = (lane < 16) ? 0 : 8;
    v8f acc = {0.f, 0.f, 0.f, 0.f, 0.f, 0.f, 0.f, 0.f};
    const unsigned short* inr = hB + (ms * 16 + mrow) * HPAD;
    const unsigned short* wp  = WpOut + ((size_t)nt * (HID / 32) * 32 + lane) * 16;
    for (int kt = 0; kt < HID / 32; ++kt) {
      BfFrag a, b;
      const v4i_ty* ar = (const v4i_ty*)(inr + kt * 32 + kl);
      a.q[0] = ar[0];
      a.q[1] = ar[2];
      const v4i_ty* w4 = (const v4i_ty*)(wp + (size_t)kt * 512);
      b.q[0] = w4[0];
      b.q[1] = w4[1];
      acc = __builtin_amdgcn_wmma_f32_16x16x32_bf16(
          false, a.v, false, b.v, (short)0, acc, false, false);
    }
    const int n  = nt * 16 + (lane & 15);
    const float bvv = b_out[n];
    const int ml = ms * 16 + ((lane < 16) ? 0 : 8);
#pragma unroll
    for (int v = 0; v < 8; ++v) {
      int rg = rowBase + ml + v;
      int j  = nbr[rg];
      float tv = tens[(size_t)j * DR + n];
      outp[(size_t)rg * DR + n] = tv - (acc[v] + bvv);
    }
  }
}

// ---------------------------------------------------------------------------
// Host launcher
// ---------------------------------------------------------------------------
extern "C" void kernel_launch(void* const* d_in, const int* in_sizes, int n_in,
                              void* d_out, int out_size, void* d_ws, size_t ws_size,
                              hipStream_t stream) {
  (void)in_sizes; (void)n_in; (void)out_size; (void)ws_size;
  const float* cond  = (const float*)d_in[0];
  const float* tens  = (const float*)d_in[1];
  const float* Win   = (const float*)d_in[2];
  const float* b_in  = (const float*)d_in[3];
  const float* W1    = (const float*)d_in[4];
  const float* b1    = (const float*)d_in[5];
  const float* W2    = (const float*)d_in[6];
  const float* b2    = (const float*)d_in[7];
  const float* W3    = (const float*)d_in[8];
  const float* b3    = (const float*)d_in[9];
  const float* Wout  = (const float*)d_in[10];
  const float* b_out = (const float*)d_in[11];
  float* outp = (float*)d_out;

  char* ws = (char*)d_ws;
  size_t off = 0;
  auto carve = [&](size_t bytes) -> char* {
    char* p = ws + off;
    off += (bytes + 255) & ~(size_t)255;
    return p;
  };
  int* nbr = (int*)carve((size_t)NPTS * KNN * sizeof(int));
  unsigned short* Xg    = (unsigned short*)carve((size_t)MROWS * DIN * 2);
  unsigned short* WpIn  = (unsigned short*)carve((size_t)(DIN / 32) * (HID / 16) * 512 * 2);
  unsigned short* Wp1   = (unsigned short*)carve((size_t)(HID / 32) * (HID / 16) * 512 * 2);
  unsigned short* Wp2   = (unsigned short*)carve((size_t)(HID / 32) * (HID / 16) * 512 * 2);
  unsigned short* Wp3   = (unsigned short*)carve((size_t)(HID / 32) * (HID / 16) * 512 * 2);
  unsigned short* WpOut = (unsigned short*)carve((size_t)(HID / 32) * (DR / 16) * 512 * 2);
  unsigned short* probeDst = (unsigned short*)carve(1024 * 2);

  knn_topk_kernel<<<NPTS, 128, 0, stream>>>(cond, nbr);
  gather_pack_x<<<((long)MROWS * DIN + 255) / 256, 256, 0, stream>>>(cond, tens, nbr, Xg);
  async_stage_probe<<<1, 128, 0, stream>>>(Xg, probeDst);
  pack_weights<<<(DIN / 32) * (HID / 16), 32, 0, stream>>>(Win,  WpIn,  DIN, HID);
  pack_weights<<<(HID / 32) * (HID / 16), 32, 0, stream>>>(W1,   Wp1,   HID, HID);
  pack_weights<<<(HID / 32) * (HID / 16), 32, 0, stream>>>(W2,   Wp2,   HID, HID);
  pack_weights<<<(HID / 32) * (HID / 16), 32, 0, stream>>>(W3,   Wp3,   HID, HID);
  pack_weights<<<(HID / 32) * (DR / 16),  32, 0, stream>>>(Wout, WpOut, HID, DR);

  const size_t smem = (size_t)(MT * XPAD + 2 * MT * HPAD) * 2;  // ~290 KB (<320 KB WGP LDS)
  fused_mlp_kernel<<<MROWS / MT, 256, smem, stream>>>(
      Xg, WpIn, Wp1, Wp2, Wp3, WpOut, b_in, b1, b2, b3, b_out, nbr, tens, outp);
}
